// EncoderLayer_16269336117672
// MI455X (gfx1250) — matrix operand order repaired and checked by reading, not compiled, not run on hardware
//
#include <hip/hip_runtime.h>
#include <hip/hip_bf16.h>

typedef __attribute__((ext_vector_type(16))) _Float16 v16h;
typedef __attribute__((ext_vector_type(8)))  float    v8f;
typedef __attribute__((ext_vector_type(4)))  unsigned int v4u;
typedef __attribute__((ext_vector_type(8)))  int      v8i_t;
typedef __attribute__((ext_vector_type(4)))  int      v4i_t;

#define D_MODEL 1024
#define N_HEADS 16
#define D_FF    4096
#define SEQ     1024
#define BATCH   4
#define M_ROWS  (BATCH * SEQ)   // 4096
#define DK      64

// ---------------------------------------------------------------------------
// WMMA operand packing helpers (CDNA5 ISA 7.12.2, wave32):
//  A (16-bit, 16x32): lane holds row m = lane%16; element e maps to
//      k = e + (e>=8 ? 8 : 0) + (lane>=16 ? 8 : 0)
//  B (16-bit, 32x16): lane holds column n = lane%16; element e maps to k as for A
// Blocks of 512 halfs stored as [tileIdx][lane][e], tileIdx = mT*kTiles + kT.
// ---------------------------------------------------------------------------
__device__ __forceinline__ size_t apack_idx(int m, int k, int kTiles) {
  int kk = k & 31;
  int hi = (kk >> 3) & 1;                       // lanes 16..31 half
  int e  = (kk & 7) | (((kk >> 4) & 1) << 3);   // element within v16h
  return ((size_t)((m >> 4) * kTiles + (k >> 5)) << 9)
       + (size_t)((((m & 15) | (hi << 4)) << 4) + e);
}

__device__ __forceinline__ size_t bpack_idx(int n, int k, int kTiles) {
  int kk = k & 31;
  int hi = (kk >> 3) & 1;
  int e  = (kk & 7) | (((kk >> 4) & 1) << 3);
  return ((size_t)((n >> 4) * kTiles + (k >> 5)) << 9)
       + (size_t)((((n & 15) | (hi << 4)) << 4) + e);
}

// ---------------------------------------------------------------------------
// TDM: DMA a 2D B-panel chunk (4 rows x 4KB, row pitch = kT*1KB) into LDS.
// D# per CDNA5 ISA; data_size = 8B units; type=2.
// Issued by wave 0 only; tracked with TENSORcnt.
// This toolchain exposes the 6-arg builtin (g0, g1, g2, g3, g_extra, cpol).
// ---------------------------------------------------------------------------
__device__ __forceinline__ void tdm_load_b_chunk(const _Float16* gsrc,
                                                 unsigned ldsByteAddr,
                                                 int rowStride8B) {
  unsigned long long ga = (unsigned long long)(size_t)gsrc;
  v4u g0;
  g0.x = 1u;                                                  // count=1 (valid user D#)
  g0.y = ldsByteAddr;                                         // lds_addr (bytes)
  g0.z = (unsigned)ga;                                        // global_addr[31:0]
  g0.w = (unsigned)((ga >> 32) & 0x01FFFFFFu) | (2u << 30);   // addr[56:32] | type=2
  v8i_t g1;
  g1[0] = (3 << 16);          // workgroup_mask=0, data_size=3 (8B), no iter/pad
  g1[1] = (512 << 16);        // atomic_barrier_addr=0 | tensor_dim0[15:0]=512
  g1[2] = (4 << 16);          // tensor_dim0[31:16]=0 | tensor_dim1[15:0]=4
  g1[3] = (512 << 16);        // tensor_dim1[31:16]=0 | tile_dim0=512 (8B units)
  g1[4] = 4;                  // tile_dim1=4 | tile_dim2=0
  g1[5] = rowStride8B;        // tensor_dim0_stride[31:0] (8B units)
  g1[6] = 0;                  // dim0_stride[47:32]=0 | dim1_stride[15:0]=0
  g1[7] = 0;
  v4i_t z4 = {0, 0, 0, 0};    // groups 2/3 unused (2D tensor)
  v8i_t z8 = {0, 0, 0, 0, 0, 0, 0, 0};
  __builtin_amdgcn_tensor_load_to_lds(g0, g1, z4, z4, z8, 0);
}

// ---------------------------------------------------------------------------
// Pack fp32 row-major [M x K] (ld = row stride) into A-operand layout (f16).
// ---------------------------------------------------------------------------
__global__ __launch_bounds__(256) void pack_a_kernel(
    const float* __restrict__ src, _Float16* __restrict__ dst,
    int mTiles, int kTiles, int ld)
{
  int t = blockIdx.x * 256 + threadIdx.x;
  int lane = t & 31, blk = t >> 5;
  if (blk >= mTiles * kTiles) return;
  int mT  = blk / kTiles;
  int kt  = blk - mT * kTiles;
  int m   = mT * 16 + (lane & 15);
  int khi = (lane >> 4) << 3;
  v16h o;
#pragma unroll
  for (int e = 0; e < 16; ++e) {
    int k = kt * 32 + e + ((e >= 8) ? 8 : 0) + khi;
    o[e] = (_Float16)src[(size_t)m * ld + k];
  }
  *(v16h*)(dst + ((size_t)blk << 9) + lane * 16) = o;
}

// Pack fp32 weight [N x K] row-major into B-operand layout: B[k][n] = W[n][k].
__global__ __launch_bounds__(256) void pack_b_kernel(
    const float* __restrict__ W, _Float16* __restrict__ dst,
    int nTiles, int kTiles, int ld)
{
  int t = blockIdx.x * 256 + threadIdx.x;
  int lane = t & 31, blk = t >> 5;
  if (blk >= nTiles * kTiles) return;
  int nT = blk / kTiles;
  int kt = blk - nT * kTiles;
  v16h o;
#pragma unroll
  for (int e = 0; e < 16; ++e)
    o[e] = (_Float16)W[(size_t)(nT * 16 + (lane & 15)) * ld + kt * 32 + e + ((e >= 8) ? 8 : 0) + ((lane >> 4) << 3)];
  *(v16h*)(dst + ((size_t)blk << 9) + lane * 16) = o;
}

// ---------------------------------------------------------------------------
// Generic WMMA GEMM: C[M,N] = Apk[M,K] * Bpk(=W^T)[K,N], f32 accumulate.
// Block = 128 threads (4 waves, 2x2); wave tile = 64x32 (4x2 WMMA tiles).
// B panel staged through LDS by the Tensor Data Mover, double-buffered
// (chunk = 4 k-steps = 16KB); A streamed from global with prefetch.
// Epilogue modes:
//   0: fp32 row-major            1: f16 A-packed (next GEMM operand)
//   2: Q  -> per-head A-pack     3: K -> per-head B-pack (K^T operand)
//   4: V  -> f16 [b*H+h][s][dk]
// ---------------------------------------------------------------------------
__global__ __launch_bounds__(128) void gemm_wmma_kernel(
    const _Float16* __restrict__ Apk, const _Float16* __restrict__ Bpk,
    int M, int N, int K,
    const float* __restrict__ bias, int relu, int mode,
    void* __restrict__ out)
{
  __shared__ alignas(16) _Float16 Bs[2][4 * 4 * 512];   // 2 x 16KB B panels
  const int lane = threadIdx.x & 31;
  const int wave = threadIdx.x >> 5;
  const int wm = wave >> 1, wn = wave & 1;
  const int kT = K >> 5;
  const int nChunks = kT >> 2;                 // 4 k-steps per chunk
  const int mTile0 = blockIdx.x * 8 + wm * 4;
  const int nTile0 = blockIdx.y * 4 + wn * 2;
  const int nTileBlk = blockIdx.y * 4;

  v8f acc[4][2] = {};
  const _Float16* aB = Apk + ((size_t)mTile0 * kT << 9) + lane * 16;
  const _Float16* bPanel = Bpk + ((size_t)nTileBlk * kT << 9);
  const int rowStride8B = kT << 7;             // kT*1024 bytes / 8

  if (threadIdx.x < 32) {
    tdm_load_b_chunk(bPanel, (unsigned)(size_t)&Bs[0][0], rowStride8B);
    __builtin_amdgcn_s_wait_tensorcnt(0);
  }
  __syncthreads();

  for (int c = 0; c < nChunks; ++c) {
    const int buf = c & 1;
    if (threadIdx.x < 32 && (c + 1) < nChunks)
      tdm_load_b_chunk(bPanel + ((size_t)((c + 1) * 4) << 9),
                       (unsigned)(size_t)&Bs[buf ^ 1][0], rowStride8B);
#pragma unroll
    for (int ktl = 0; ktl < 4; ++ktl) {
      const int kt = c * 4 + ktl;
      if (kt + 1 < kT)
        __builtin_prefetch(aB + ((size_t)(kt + 1) << 9), 0, 1);   // global_prefetch_b8
      v16h a[4], b[2];
#pragma unroll
      for (int i = 0; i < 4; ++i)
        a[i] = *(const v16h*)(aB + (((size_t)i * kT + kt) << 9));
#pragma unroll
      for (int j = 0; j < 2; ++j)
        b[j] = *(const v16h*)(&Bs[buf][((wn * 2 + j) * 4 + ktl) * 512 + lane * 16]);
#pragma unroll
      for (int i = 0; i < 4; ++i)
#pragma unroll
        for (int j = 0; j < 2; ++j)
          acc[i][j] = __builtin_amdgcn_wmma_f32_16x16x32_f16(
              false, a[i], false, b[j], (short)0, acc[i][j], false, false);
    }
    if (threadIdx.x < 32) __builtin_amdgcn_s_wait_tensorcnt(0);
    __syncthreads();
  }

  const int rowAdd = (lane >> 4) << 3;   // C layout: row = r + 8*(lane/16)
  const int cl = lane & 15;              //           col = lane%16
#pragma unroll
  for (int i = 0; i < 4; ++i) {
#pragma unroll
    for (int j = 0; j < 2; ++j) {
#pragma unroll
      for (int r = 0; r < 8; ++r) {
        int m = (mTile0 + i) * 16 + r + rowAdd;
        int n = (nTile0 + j) * 16 + cl;
        float v = acc[i][j][r];
        if (bias) v += bias[n];
        if (relu) v = fmaxf(v, 0.0f);
        if (mode == 0) {
          ((float*)out)[(size_t)m * N + n] = v;
        } else if (mode == 1) {
          ((_Float16*)out)[apack_idx(m, n, N >> 5)] = (_Float16)v;
        } else if (mode == 2) {
          int bi = m >> 10, s = m & 1023, h = n >> 6, c2 = n & 63;
          size_t base = (size_t)(bi * N_HEADS + h) * (64 * 2 * 512);
          ((_Float16*)out)[base + apack_idx(s, c2, 2)] = (_Float16)v;
        } else if (mode == 3) {
          int bi = m >> 10, s = m & 1023, h = n >> 6, c2 = n & 63;
          size_t base = (size_t)(bi * N_HEADS + h) * (64 * 2 * 512);
          ((_Float16*)out)[base + bpack_idx(s, c2, 2)] = (_Float16)v;
        } else {               // mode 4: V as f16 [bh][s][dk]
          int bi = m >> 10, s = m & 1023, h = n >> 6, c2 = n & 63;
          ((_Float16*)out)[((size_t)(bi * N_HEADS + h) * SEQ + s) * DK + c2] =
              (_Float16)v;
        }
      }
    }
  }
}

// ---------------------------------------------------------------------------
// Attention: one wave32 per (query 16-tile, b*H+h).
// scores(16x1024, fp32) in exactly 64KB LDS; QK^T and P*V via WMMA.
// ---------------------------------------------------------------------------
__global__ __launch_bounds__(32) void attn_kernel(
    const _Float16* __restrict__ Qpk,  // [BH][64*2*512] A-pack (S x dk)
    const _Float16* __restrict__ KTpk, // [BH][64*2*512] B-pack (K^T operand)
    const _Float16* __restrict__ Vh,   // [BH][SEQ][DK] f16
    const float* __restrict__ rel_table,  // [255][N_HEADS]
    _Float16* __restrict__ ctxApk)        // A-pack of ctx [M_ROWS x D_MODEL]
{
  __shared__ float sc[16 * SEQ];       // 64 KB
  const int lane = threadIdx.x;
  const int qT = blockIdx.x;           // 0..63
  const int bh = blockIdx.y;           // 0..63
  const int h  = bh & (N_HEADS - 1);
  const int rowAdd = (lane >> 4) << 3;
  const int cl = lane & 15;

  const _Float16* qb = Qpk  + (size_t)bh * (128 * 512);
  const _Float16* kb = KTpk + (size_t)bh * (128 * 512);
  v16h aq0 = *(const v16h*)(qb + ((size_t)(qT * 2 + 0) << 9) + lane * 16);
  v16h aq1 = *(const v16h*)(qb + ((size_t)(qT * 2 + 1) << 9) + lane * 16);

  // ---- scores = Q K^T / sqrt(dk) + rel_bias -> LDS
  for (int nt = 0; nt < 64; ++nt) {
    v16h b0 = *(const v16h*)(kb + ((size_t)(nt * 2 + 0) << 9) + lane * 16);
    v16h b1 = *(const v16h*)(kb + ((size_t)(nt * 2 + 1) << 9) + lane * 16);
    v8f s = {};
    s = __builtin_amdgcn_wmma_f32_16x16x32_f16(false, aq0, false, b0, (short)0, s, false, false);
    s = __builtin_amdgcn_wmma_f32_16x16x32_f16(false, aq1, false, b1, (short)0, s, false, false);
#pragma unroll
    for (int r = 0; r < 8; ++r) {
      int row = r + rowAdd;
      int q = qT * 16 + row;
      int k = nt * 16 + cl;
      int ri = q - k + 127;
      ri = (ri < 0) ? 0 : (ri > 254 ? 254 : ri);
      sc[row * SEQ + k] = s[r] * 0.125f + rel_table[ri * N_HEADS + h];
    }
  }
  __syncthreads();

  // ---- row-wise softmax (lane-cooperative, wave32 shfl reductions)
  for (int row = 0; row < 16; ++row) {
    float mx = -3.402823466e38f;
    for (int j = lane; j < SEQ; j += 32) mx = fmaxf(mx, sc[row * SEQ + j]);
#pragma unroll
    for (int o = 16; o; o >>= 1) mx = fmaxf(mx, __shfl_xor(mx, o, 32));
    float sum = 0.f;
    for (int j = lane; j < SEQ; j += 32) {
      float e = __expf(sc[row * SEQ + j] - mx);
      sc[row * SEQ + j] = e;
      sum += e;
    }
#pragma unroll
    for (int o = 16; o; o >>= 1) sum += __shfl_xor(sum, o, 32);
    float invs = 1.0f / sum;
    for (int j = lane; j < SEQ; j += 32) sc[row * SEQ + j] *= invs;
  }
  __syncthreads();

  // ---- ctx = P * V  (A frag built from LDS, B = f16 V rows, col n = lane%16)
  v8f acc[4] = {};
  const _Float16* vb = Vh + (size_t)bh * SEQ * DK;
  const int arow = lane & 15;
  const int khi = (lane >> 4) << 3;
  for (int kt = 0; kt < 32; ++kt) {
    v16h pa;
#pragma unroll
    for (int e = 0; e < 16; ++e) {
      int k = kt * 32 + e + ((e >= 8) ? 8 : 0) + khi;
      pa[e] = (_Float16)sc[arow * SEQ + k];
    }
    const _Float16* vrow = vb + (size_t)(kt * 32 + khi) * DK + arow;
#pragma unroll
    for (int ntc = 0; ntc < 4; ++ntc) {
      v16h bv;
#pragma unroll
      for (int e = 0; e < 16; ++e)
        bv[e] = vrow[(size_t)(e + ((e >= 8) ? 8 : 0)) * DK + ntc * 16];
      acc[ntc] = __builtin_amdgcn_wmma_f32_16x16x32_f16(
          false, pa, false, bv, (short)0, acc[ntc], false, false);
    }
  }

  // ---- write ctx directly in A-pack layout for the Wo GEMM
  const int bi = bh >> 4;
#pragma unroll
  for (int ntc = 0; ntc < 4; ++ntc) {
#pragma unroll
    for (int r = 0; r < 8; ++r) {
      int q = qT * 16 + r + rowAdd;
      int c = ntc * 16 + cl;
      int m = bi * SEQ + q;
      int k = h * DK + c;
      ctxApk[apack_idx(m, k, D_MODEL >> 5)] = (_Float16)acc[ntc][r];
    }
  }
}

// ---------------------------------------------------------------------------
// out = LayerNorm(a + b) * g + beta ; optionally also emit A-packed f16 copy.
// ---------------------------------------------------------------------------
__global__ __launch_bounds__(256) void add_ln_kernel(
    const float* __restrict__ a, const float* __restrict__ b,
    const float* __restrict__ g, const float* __restrict__ beta,
    float* __restrict__ outF, _Float16* __restrict__ outPk)
{
  const int D = D_MODEL;
  int row = blockIdx.x, tid = threadIdx.x;
  __shared__ float red[8];
  float v[4];
  float s = 0.f;
#pragma unroll
  for (int i = 0; i < 4; ++i) {
    int c = tid + i * 256;
    v[i] = a[(size_t)row * D + c] + b[(size_t)row * D + c];
    s += v[i];
  }
#pragma unroll
  for (int o = 16; o; o >>= 1) s += __shfl_xor(s, o, 32);
  if ((tid & 31) == 0) red[tid >> 5] = s;
  __syncthreads();
  float mu = 0.f;
#pragma unroll
  for (int i = 0; i < 8; ++i) mu += red[i];
  mu *= (1.0f / D);
  __syncthreads();
  float s2 = 0.f;
#pragma unroll
  for (int i = 0; i < 4; ++i) { float d = v[i] - mu; s2 += d * d; }
#pragma unroll
  for (int o = 16; o; o >>= 1) s2 += __shfl_xor(s2, o, 32);
  if ((tid & 31) == 0) red[tid >> 5] = s2;
  __syncthreads();
  float var = 0.f;
#pragma unroll
  for (int i = 0; i < 8; ++i) var += red[i];
  var *= (1.0f / D);
  float inv = rsqrtf(var + 1e-5f);
#pragma unroll
  for (int i = 0; i < 4; ++i) {
    int c = tid + i * 256;
    float o = (v[i] - mu) * inv * g[c] + beta[c];
    if (outF)  outF[(size_t)row * D + c] = o;
    if (outPk) outPk[apack_idx(row, c, D >> 5)] = (_Float16)o;
  }
}

// ---------------------------------------------------------------------------
extern "C" void kernel_launch(void* const* d_in, const int* in_sizes, int n_in,
                              void* d_out, int out_size, void* d_ws, size_t ws_size,
                              hipStream_t stream) {
  (void)in_sizes; (void)n_in; (void)out_size; (void)ws_size;
  const float* x   = (const float*)d_in[0];
  const float* Wq  = (const float*)d_in[1];
  const float* Wk  = (const float*)d_in[2];
  const float* Wv  = (const float*)d_in[3];
  const float* Wo  = (const float*)d_in[4];
  const float* bo  = (const float*)d_in[5];
  const float* rel = (const float*)d_in[6];
  const float* W1  = (const float*)d_in[7];
  const float* b1  = (const float*)d_in[8];
  const float* W2  = (const float*)d_in[9];
  const float* b2  = (const float*)d_in[10];
  const float* g1  = (const float*)d_in[11];
  const float* be1 = (const float*)d_in[12];
  const float* g2  = (const float*)d_in[13];
  const float* be2 = (const float*)d_in[14];

  char* ws = (char*)d_ws;
  size_t off = 0;
  auto take = [&](size_t bytes) -> void* {
    void* p = ws + off;
    off += (bytes + 255) & ~(size_t)255;
    return p;
  };
  _Float16* xApk   = (_Float16*)take((size_t)M_ROWS * D_MODEL * 2);
  _Float16* WqPk   = (_Float16*)take((size_t)D_MODEL * D_MODEL * 2);
  _Float16* WkPk   = (_Float16*)take((size_t)D_MODEL * D_MODEL * 2);
  _Float16* WvPk   = (_Float16*)take((size_t)D_MODEL * D_MODEL * 2);
  _Float16* WoPk   = (_Float16*)take((size_t)D_MODEL * D_MODEL * 2);
  _Float16* W1Pk   = (_Float16*)take((size_t)D_FF * D_MODEL * 2);
  _Float16* W2Pk   = (_Float16*)take((size_t)D_MODEL * D_FF * 2);
  _Float16* Qpk    = (_Float16*)take((size_t)64 * 128 * 512 * 2);
  _Float16* KTpk   = (_Float16*)take((size_t)64 * 128 * 512 * 2);
  _Float16* Vh     = (_Float16*)take((size_t)64 * SEQ * DK * 2);
  _Float16* ctxApk = (_Float16*)take((size_t)M_ROWS * D_MODEL * 2);
  float*    attnO  = (float*)   take((size_t)M_ROWS * D_MODEL * 4);
  float*    hRes   = (float*)   take((size_t)M_ROWS * D_MODEL * 4);
  _Float16* hApk   = (_Float16*)take((size_t)M_ROWS * D_MODEL * 2);
  _Float16* y1Apk  = (_Float16*)take((size_t)M_ROWS * D_FF * 2);
  float*    y2     = (float*)   take((size_t)M_ROWS * D_MODEL * 4);

  // --- operand packing
  {
    int threads = 256 * 32 * 32;   // mTiles=256, kTiles=32
    pack_a_kernel<<<threads / 256, 256, 0, stream>>>(x, xApk, 256, 32, D_MODEL);
  }
  auto packB = [&](const float* W, _Float16* dst, int Nt, int kT, int ld) {
    int threads = Nt * kT * 32;
    pack_b_kernel<<<(threads + 255) / 256, 256, 0, stream>>>(W, dst, Nt, kT, ld);
  };
  packB(Wq, WqPk, 64, 32, D_MODEL);
  packB(Wk, WkPk, 64, 32, D_MODEL);
  packB(Wv, WvPk, 64, 32, D_MODEL);
  packB(Wo, WoPk, 64, 32, D_MODEL);
  packB(W1, W1Pk, 256, 32, D_MODEL);
  packB(W2, W2Pk, 64, 128, D_FF);

  dim3 blk(128), g16(32, 16), g64(32, 64);
  // QKV projections: epilogues repack for attention directly
  gemm_wmma_kernel<<<g16, blk, 0, stream>>>(xApk, WqPk, M_ROWS, D_MODEL, D_MODEL, nullptr, 0, 2, Qpk);
  gemm_wmma_kernel<<<g16, blk, 0, stream>>>(xApk, WkPk, M_ROWS, D_MODEL, D_MODEL, nullptr, 0, 3, KTpk);
  gemm_wmma_kernel<<<g16, blk, 0, stream>>>(xApk, WvPk, M_ROWS, D_MODEL, D_MODEL, nullptr, 0, 4, Vh);

  attn_kernel<<<dim3(64, 64), 32, 0, stream>>>(Qpk, KTpk, Vh, rel, ctxApk);

  gemm_wmma_kernel<<<g16, blk, 0, stream>>>(ctxApk, WoPk, M_ROWS, D_MODEL, D_MODEL, bo, 0, 0, attnO);
  add_ln_kernel<<<M_ROWS, 256, 0, stream>>>(x, attnO, g1, be1, hRes, hApk);

  gemm_wmma_kernel<<<g64, blk, 0, stream>>>(hApk, W1Pk, M_ROWS, D_FF, D_MODEL, b1, 1, 1, y1Apk);
  gemm_wmma_kernel<<<g16, blk, 0, stream>>>(y1Apk, W2Pk, M_ROWS, D_MODEL, D_FF, b2, 0, 0, y2);
  add_ln_kernel<<<M_ROWS, 256, 0, stream>>>(hRes, y2, g2, be2, (float*)d_out, nullptr);
}
